// Attention_19653770346940
// MI455X (gfx1250) — compile-verified, matrix operand order checked by reading, not run on hardware
//
#include <hip/hip_runtime.h>
#include <stdint.h>

typedef __bf16 bf16;
typedef __attribute__((ext_vector_type(16))) __bf16 v16bf;
typedef __attribute__((ext_vector_type(8)))  __bf16 bf16x8;
typedef __attribute__((ext_vector_type(4)))  __bf16 bf16x4;
typedef __attribute__((ext_vector_type(8)))  float  v8f;
typedef __attribute__((ext_vector_type(4)))  int    v4i;

#define B_   2
#define S_   2048
#define H_   16
#define D_   64
#define DIM_ 1024
#define NINF (-10000.0f)

#ifndef __has_builtin
#define __has_builtin(x) 0
#endif

#if __has_builtin(__builtin_amdgcn_global_load_async_to_lds_b128)
#define HAVE_ASYNC 1
#else
#define HAVE_ASYNC 0
#endif

#if HAVE_ASYNC
#define AS1 __attribute__((address_space(1)))
#define AS3 __attribute__((address_space(3)))
// param1: v4i in AS1 (printed as "__device__" by clang); param2: v4i in AS3.
// flat LDS address low 32 bits == LDS byte offset (aperture rules); flat==global for AS1.
__device__ inline void async_copy16(const void* g, void* l) {
  __builtin_amdgcn_global_load_async_to_lds_b128(
      (AS1 v4i*)(unsigned long long)(uintptr_t)g,
      (AS3 v4i*)(unsigned int)(uintptr_t)l, 0, 0);
}
#endif

__device__ inline void wait_async() {
#if HAVE_ASYNC
#if __has_builtin(__builtin_amdgcn_s_wait_asynccnt)
  __builtin_amdgcn_s_wait_asynccnt(0);
#else
  asm volatile("s_wait_asynccnt 0x0" ::: "memory");
#endif
#endif
}

__device__ inline v8f wmma_bf16(v16bf a, v16bf b, v8f c) {
  return __builtin_amdgcn_wmma_f32_16x16x32_bf16(false, a, false, b, (short)0, c, false, false);
}

// ---------------------------------------------------------------- convert
__global__ __launch_bounds__(256) void cvt_f32_bf16(const float* __restrict__ in,
                                                    bf16* __restrict__ out, int n) {
  int i = blockIdx.x * 256 + threadIdx.x;
  if (i < n) out[i] = (bf16)in[i];
}

// ---------------------------------------------------------------- layernorm
__global__ __launch_bounds__(256) void layernorm_bf16(const float* __restrict__ x,
                                                      const float* __restrict__ g,
                                                      const float* __restrict__ b,
                                                      bf16* __restrict__ h) {
  __shared__ float red[8];
  const int row = blockIdx.x;
  const int tid = threadIdx.x;
  const int wave = tid >> 5, lane = tid & 31;
  const float* xr = x + (size_t)row * DIM_;

  float v[4];
  float s = 0.f;
#pragma unroll
  for (int i = 0; i < 4; ++i) { v[i] = xr[tid + 256 * i]; s += v[i]; }
#pragma unroll
  for (int off = 16; off >= 1; off >>= 1) s += __shfl_xor(s, off, 32);
  if (lane == 0) red[wave] = s;
  __syncthreads();
  float tot = 0.f;
#pragma unroll
  for (int i = 0; i < 8; ++i) tot += red[i];
  const float mu = tot * (1.0f / DIM_);
  __syncthreads();

  float vs = 0.f;
#pragma unroll
  for (int i = 0; i < 4; ++i) { float d = v[i] - mu; vs += d * d; }
#pragma unroll
  for (int off = 16; off >= 1; off >>= 1) vs += __shfl_xor(vs, off, 32);
  if (lane == 0) red[wave] = vs;
  __syncthreads();
  float vtot = 0.f;
#pragma unroll
  for (int i = 0; i < 8; ++i) vtot += red[i];
  const float rstd = __frsqrt_rn(vtot * (1.0f / DIM_) + 1e-6f);

#pragma unroll
  for (int i = 0; i < 4; ++i) {
    int c = tid + 256 * i;
    h[(size_t)row * DIM_ + c] = (bf16)((v[i] - mu) * rstd * g[c] + b[c]);
  }
}

// ---------------------------------------------------------------- GEMM (bf16 WMMA)
// C[M,N] = A[M,K] @ B[K,N] + bias ; optional residual ; out bf16 or f32
// 128x128 block tile, BK=32, 8 waves (4x2) x (2x4) 16x16 WMMA tiles.
// Double-buffered LDS, A tile via async-to-LDS (when available), B tile
// register-staged + transposed for pair-contiguous B^T fragment reads.
#define BM 128
#define BN 128
#define BK 32
#define ASTR 56   // 112B stride: 16B aligned, 28-bank stride -> conflict free
#define BSTR 40   // 80B stride: 16B aligned, 20-bank stride -> conflict free

template<int Mi, int Ni, int Ki, bool HAS_RES, bool OUT_BF16>
__global__ __launch_bounds__(256)
void gemm_bf16_wmma(const bf16* __restrict__ A, const bf16* __restrict__ Bm,
                    const float* __restrict__ bias, const float* __restrict__ res,
                    void* __restrict__ outp) {
  __shared__ bf16 As[2][BM][ASTR];
  __shared__ bf16 BsT[2][BN][BSTR];   // BsT[n][k]

  const int tid = threadIdx.x;
  const int wave = tid >> 5, lane = tid & 31;
  const int wm = wave & 3, wn = wave >> 2;       // 4x2 wave grid
  const int hf = lane >> 4, ln16 = lane & 15;
  const int m0 = blockIdx.y * BM;
  const int n0 = blockIdx.x * BN;

  v8f acc[2][4];
#pragma unroll
  for (int i = 0; i < 2; ++i)
#pragma unroll
    for (int j = 0; j < 4; ++j) acc[i][j] = (v8f){0, 0, 0, 0, 0, 0, 0, 0};

  bf16x8 rb[2];   // staged B chunks

  auto copyA = [&](int ks, int buf) {
#pragma unroll
    for (int it = 0; it < 2; ++it) {
      int i = tid + it * 256;                 // 512 chunks of 8 halfs
      int r = i >> 2, c8 = (i & 3) * 8;
      const bf16* gp = &A[(size_t)(m0 + r) * Ki + ks + c8];
      bf16* lp = &As[buf][r][c8];
#if HAVE_ASYNC
      async_copy16(gp, lp);
#else
      *(bf16x8*)lp = *(const bf16x8*)gp;
#endif
    }
  };
  auto loadB = [&](int ks) {
#pragma unroll
    for (int it = 0; it < 2; ++it) {
      int i = tid + it * 256;
      int k = i >> 4, c8 = (i & 15) * 8;
      rb[it] = *(const bf16x8*)&Bm[(size_t)(ks + k) * Ni + n0 + c8];
    }
  };
  auto storeB = [&](int buf) {
#pragma unroll
    for (int it = 0; it < 2; ++it) {
      int i = tid + it * 256;
      int k = i >> 4, c8 = (i & 15) * 8;
#pragma unroll
      for (int j = 0; j < 8; ++j) BsT[buf][c8 + j][k] = rb[it][j];
    }
  };

  copyA(0, 0);    // prime pipeline
  loadB(0);
  constexpr int NST = Ki / BK;
#pragma unroll 1
  for (int s = 0; s < NST; ++s) {
    const int cur = s & 1;
    storeB(cur);
    wait_async();
    __syncthreads();
    if (s + 1 < NST) {          // overlap next tile's loads with compute
      copyA((s + 1) * BK, cur ^ 1);
      loadB((s + 1) * BK);
    }

    v16bf af[2], bfr[4];
#pragma unroll
    for (int fm = 0; fm < 2; ++fm) {
      int row = wm * 32 + fm * 16 + ln16;
#pragma unroll
      for (int r = 0; r < 8; ++r) {
        int kk = ((r & 3) * 2) + ((r >> 2) * 16) + hf * 8;
        af[fm][2 * r]     = As[cur][row][kk];
        af[fm][2 * r + 1] = As[cur][row][kk + 1];
      }
    }
#pragma unroll
    for (int fn = 0; fn < 4; ++fn) {
      int col = wn * 64 + fn * 16 + ln16;
#pragma unroll
      for (int r = 0; r < 8; ++r) {
        int kk = ((r & 3) * 2) + ((r >> 2) * 16) + hf * 8;
        bfr[fn][2 * r]     = BsT[cur][col][kk];
        bfr[fn][2 * r + 1] = BsT[cur][col][kk + 1];
      }
    }
#pragma unroll
    for (int fm = 0; fm < 2; ++fm)
#pragma unroll
      for (int fn = 0; fn < 4; ++fn)
        acc[fm][fn] = wmma_bf16(af[fm], bfr[fn], acc[fm][fn]);
  }

  // straight-line epilogue (compile-time N, residual, output type)
#pragma unroll
  for (int fm = 0; fm < 2; ++fm)
#pragma unroll
    for (int fn = 0; fn < 4; ++fn)
#pragma unroll
      for (int v = 0; v < 8; ++v) {
        int row = m0 + wm * 32 + fm * 16 + v + hf * 8;
        int col = n0 + wn * 64 + fn * 16 + ln16;
        size_t idx = (size_t)row * Ni + col;
        float val = acc[fm][fn][v] + bias[col];
        if constexpr (HAS_RES) val += res[idx];
        if constexpr (OUT_BF16) ((bf16*)outp)[idx] = (bf16)val;
        else                    ((float*)outp)[idx] = val;
      }
}

// ---------------------------------------------------------------- flash attention
// block = 64 queries of one (b,h); 4 waves x 16 query rows
__global__ __launch_bounds__(128)
void attn_flash(const bf16* __restrict__ qkv, bf16* __restrict__ ctx) {
  __shared__ bf16 Qs[64][72];    // [q][d], pre-scaled
  __shared__ bf16 Ks[64][72];    // [key][d] (natural B^T layout for Q.K^T)
  __shared__ bf16 VsT[64][72];   // [d][key] (pair-contiguous B reads for P.V)
  __shared__ bf16 Ps[64][72];    // [q][key]

  const int blk = blockIdx.x;
  const int qt = blk & 31;
  const int bh = blk >> 5;
  const int h = bh & 15, b = bh >> 4;
  const int q0 = qt * 64;
  const int tid = threadIdx.x;
  const int wave = tid >> 5, lane = tid & 31;
  const int hf = lane >> 4, ln16 = lane & 15;
  const float scale = 0.125f;   // 1/sqrt(64)

  // load Q (scaled, bf16)
#pragma unroll
  for (int it = 0; it < 4; ++it) {
    int i = tid + it * 128;              // 512 chunks of 8
    int r = i >> 3, c8 = (i & 7) * 8;
    bf16x8 d = *(const bf16x8*)&qkv[(size_t)(b * S_ + q0 + r) * 3072 + h * 64 + c8];
    bf16x4 lo, hi2;
#pragma unroll
    for (int j = 0; j < 4; ++j) lo[j]  = (bf16)((float)d[j] * scale);
#pragma unroll
    for (int j = 0; j < 4; ++j) hi2[j] = (bf16)((float)d[j + 4] * scale);
    *(bf16x4*)&Qs[r][c8]     = lo;
    *(bf16x4*)&Qs[r][c8 + 4] = hi2;
  }

  v8f o[4];
#pragma unroll
  for (int t = 0; t < 4; ++t) o[t] = (v8f){0, 0, 0, 0, 0, 0, 0, 0};
  float mrow[8], lrow[8];
#pragma unroll
  for (int v = 0; v < 8; ++v) { mrow[v] = -1e30f; lrow[v] = 0.f; }

  for (int kt = 0; kt <= qt; ++kt) {
    __syncthreads();     // previous-iter readers done (also orders Q store 1st iter)
#pragma unroll
    for (int it = 0; it < 4; ++it) {
      int i = tid + it * 128;
      int r = i >> 3, c8 = (i & 7) * 8;
      size_t base = (size_t)(b * S_ + kt * 64 + r) * 3072 + h * 64 + c8;
#if HAVE_ASYNC
      async_copy16(&qkv[base + 1024], &Ks[r][c8]);   // K tile: ASYNCcnt path
#else
      bf16x8 kd = *(const bf16x8*)&qkv[base + 1024];
      *(bf16x4*)&Ks[r][c8]     = (bf16x4){kd[0], kd[1], kd[2], kd[3]};
      *(bf16x4*)&Ks[r][c8 + 4] = (bf16x4){kd[4], kd[5], kd[6], kd[7]};
#endif
      bf16x8 vd = *(const bf16x8*)&qkv[base + 2048];
#pragma unroll
      for (int j = 0; j < 8; ++j) VsT[c8 + j][r] = vd[j];
    }
    wait_async();
    __syncthreads();

    // scores S = Qw (16x64) @ K^T : 4 N-tiles, 2 K-steps each
    v8f sc[4];
#pragma unroll
    for (int t = 0; t < 4; ++t) {
      v8f a = (v8f){0, 0, 0, 0, 0, 0, 0, 0};
#pragma unroll
      for (int ks = 0; ks < 64; ks += 32) {
        v16bf af, bfr;
        int row = wave * 16 + ln16;
        int colk = t * 16 + ln16;
#pragma unroll
        for (int r = 0; r < 8; ++r) {
          int kk = ks + ((r & 3) * 2) + ((r >> 2) * 16) + hf * 8;
          af[2 * r]      = Qs[row][kk];
          af[2 * r + 1]  = Qs[row][kk + 1];
          bfr[2 * r]     = Ks[colk][kk];
          bfr[2 * r + 1] = Ks[colk][kk + 1];
        }
        a = wmma_bf16(af, bfr, a);
      }
      sc[t] = a;
    }

    // causal mask on diagonal tile (additive NINF to match reference)
    if (kt == qt) {
#pragma unroll
      for (int t = 0; t < 4; ++t)
#pragma unroll
        for (int v = 0; v < 8; ++v) {
          int qr = wave * 16 + v + hf * 8;
          int kc = t * 16 + ln16;
          if (kc > qr) sc[t][v] += NINF;
        }
    }

    // online softmax (row lives in 16 lanes of one half-wave)
#pragma unroll
    for (int v = 0; v < 8; ++v) {
      float mx = fmaxf(fmaxf(sc[0][v], sc[1][v]), fmaxf(sc[2][v], sc[3][v]));
#pragma unroll
      for (int off = 1; off < 16; off <<= 1) mx = fmaxf(mx, __shfl_xor(mx, off, 32));
      float mnew = fmaxf(mrow[v], mx);
      float corr = __expf(mrow[v] - mnew);
#pragma unroll
      for (int t = 0; t < 4; ++t) o[t][v] *= corr;
      float rsum = 0.f;
#pragma unroll
      for (int t = 0; t < 4; ++t) {
        float p = __expf(sc[t][v] - mnew);
        sc[t][v] = p;
        rsum += p;
      }
#pragma unroll
      for (int off = 1; off < 16; off <<= 1) rsum += __shfl_xor(rsum, off, 32);
      lrow[v] = lrow[v] * corr + rsum;
      mrow[v] = mnew;
    }

    // P -> LDS (per-wave private 16-row region; same-wave DS ordering suffices)
#pragma unroll
    for (int t = 0; t < 4; ++t)
#pragma unroll
      for (int v = 0; v < 8; ++v)
        Ps[wave * 16 + v + hf * 8][t * 16 + ln16] = (bf16)sc[t][v];

    // O += P (16x64) @ V (64x64)
#pragma unroll
    for (int t = 0; t < 4; ++t) {
      v8f a = o[t];
#pragma unroll
      for (int ks = 0; ks < 64; ks += 32) {
        v16bf af, bfr;
        int row = wave * 16 + ln16;
        int cold = t * 16 + ln16;
#pragma unroll
        for (int r = 0; r < 8; ++r) {
          int kk = ks + ((r & 3) * 2) + ((r >> 2) * 16) + hf * 8;
          af[2 * r]      = Ps[row][kk];
          af[2 * r + 1]  = Ps[row][kk + 1];
          bfr[2 * r]     = VsT[cold][kk];
          bfr[2 * r + 1] = VsT[cold][kk + 1];
        }
        a = wmma_bf16(af, bfr, a);
      }
      o[t] = a;
    }
  }

  // normalize + write ctx (bf16, [B*S][DIM] at col h*64)
#pragma unroll
  for (int v = 0; v < 8; ++v) {
    float inv = 1.0f / lrow[v];
    int qr = q0 + wave * 16 + v + hf * 8;
#pragma unroll
    for (int t = 0; t < 4; ++t)
      ctx[(size_t)(b * S_ + qr) * DIM_ + h * 64 + t * 16 + ln16] = (bf16)(o[t][v] * inv);
  }
}

// ---------------------------------------------------------------- host launch
static inline size_t align256(size_t x) { return (x + 255) & ~(size_t)255; }

extern "C" void kernel_launch(void* const* d_in, const int* in_sizes, int n_in,
                              void* d_out, int out_size, void* d_ws, size_t ws_size,
                              hipStream_t stream) {
  const float* x    = (const float*)d_in[0];
  const float* ln_g = (const float*)d_in[1];
  const float* ln_b = (const float*)d_in[2];
  const float* wqkv = (const float*)d_in[3];
  const float* bqkv = (const float*)d_in[4];
  const float* wo   = (const float*)d_in[5];
  const float* bo   = (const float*)d_in[6];
  float* out = (float*)d_out;

  const int M = B_ * S_;          // 4096
  const int N1 = 3 * DIM_;        // 3072
  const int K = DIM_;             // 1024

  char* ws = (char*)d_ws;
  size_t off = 0;
  bf16* wqkv_b = (bf16*)(ws + off); off = align256(off + (size_t)K * N1 * 2);
  bf16* wo_b   = (bf16*)(ws + off); off = align256(off + (size_t)K * DIM_ * 2);
  bf16* h_b    = (bf16*)(ws + off); off = align256(off + (size_t)M * DIM_ * 2);
  bf16* qkv_b  = (bf16*)(ws + off); off = align256(off + (size_t)M * N1 * 2);
  bf16* ctx_b  = (bf16*)(ws + off); off = align256(off + (size_t)M * DIM_ * 2);
  (void)ws_size; (void)in_sizes; (void)n_in; (void)out_size;

  // 1) weights -> bf16
  {
    int n = K * N1;
    cvt_f32_bf16<<<(n + 255) / 256, 256, 0, stream>>>(wqkv, wqkv_b, n);
    n = K * DIM_;
    cvt_f32_bf16<<<(n + 255) / 256, 256, 0, stream>>>(wo, wo_b, n);
  }
  // 2) layernorm
  layernorm_bf16<<<M, 256, 0, stream>>>(x, ln_g, ln_b, h_b);
  // 3) QKV GEMM: [4096,1024]x[1024,3072] + bqkv -> bf16
  gemm_bf16_wmma<4096, 3072, 1024, false, true>
      <<<dim3(N1 / BN, M / BM), 256, 0, stream>>>(h_b, wqkv_b, bqkv, nullptr, (void*)qkv_b);
  // 4) causal flash attention -> ctx bf16
  attn_flash<<<B_ * H_ * (S_ / 64), 128, 0, stream>>>(qkv_b, ctx_b);
  // 5) output GEMM + bias + residual -> f32 out
  gemm_bf16_wmma<4096, 1024, 1024, true, false>
      <<<dim3(DIM_ / BN, M / BM), 256, 0, stream>>>(ctx_b, wo_b, bo, x, (void*)out);
}
